// TriangleIndexes_11647951307124
// MI455X (gfx1250) — compile-verified, matrix operand order checked by reading, not run on hardware
//
#include <hip/hip_runtime.h>
#include <hip/hip_bf16.h>
#include <math.h>

typedef __attribute__((ext_vector_type(16))) _Float16 v16h;
typedef __attribute__((ext_vector_type(8)))  float    v8f;
typedef __attribute__((ext_vector_type(4)))  float    v4f;

#define NEI 15

// ---------------------------------------------------------------------------
// Kernel 1: extract per-row neighbor column indices from dense adjacency.
// One wave (32 lanes) per row; each lane loads a 16B-aligned float4 chunk with
// a non-temporal hint (read-once 400MB stream), ballot-compacts nonzero column
// indices in ascending column order into nb[row*16 + 0..14].
// ---------------------------------------------------------------------------
__global__ void extract_nb_kernel(const float* __restrict__ adj,
                                  int* __restrict__ nb, int N) {
    const int lane = threadIdx.x & 31;
    const int row  = blockIdx.x * (blockDim.x >> 5) + (threadIdx.x >> 5);
    if (row >= N) return;  // wave-uniform

    const float* rp = adj + (size_t)row * (size_t)N;
    const unsigned long long lt = (1ull << lane) - 1ull;
    int cnt = 0;

    for (int base = 0; base < N; base += 128) {
        // speculative prefetch one chunk ahead -> global_prefetch_b8
        __builtin_prefetch(rp + base + 128, 0, 1);

        const int c0 = base + lane * 4;   // N % 4 == 0 -> full v4f when c0 < N
        v4f v = (v4f)0.0f;
        if (c0 < N) {
            v = __builtin_nontemporal_load((const v4f*)(rp + c0));
        }
        bool p0 = (c0 < N) && (v.x != 0.0f);
        bool p1 = (c0 < N) && (v.y != 0.0f);
        bool p2 = (c0 < N) && (v.z != 0.0f);
        bool p3 = (c0 < N) && (v.w != 0.0f);

        unsigned long long b0 = __ballot(p0);
        unsigned long long b1 = __ballot(p1);
        unsigned long long b2 = __ballot(p2);
        unsigned long long b3 = __ballot(p3);

        // total nonzeros in lanes below me (column order = (lane, elem) order)
        int before = __popcll(b0 & lt) + __popcll(b1 & lt) +
                     __popcll(b2 & lt) + __popcll(b3 & lt);
        int w = cnt + before;
        if (p0) { if (w < 16) nb[row * 16 + w] = c0 + 0; w++; }
        if (p1) { if (w < 16) nb[row * 16 + w] = c0 + 1; w++; }
        if (p2) { if (w < 16) nb[row * 16 + w] = c0 + 2; w++; }
        if (p3) { if (w < 16) nb[row * 16 + w] = c0 + 3; w++; }

        cnt += __popcll(b0) + __popcll(b1) + __popcll(b2) + __popcll(b3);
    }
}

// ---------------------------------------------------------------------------
// Kernel 2: per-node triangle count via WMMA. One wave per node.
// Build a 16x16 0/1 mask E[j][k] = (j<k) && nb[j]>i && nb[k]>i && adjacent.
// Feed E (zero-padded to 16x32 f16) as the A operand of
// v_wmma_f32_16x16x32_f16 against an all-ones B with C=0: every E element
// enters exactly one (M,K) slot, so sum(D) = 16 * sum(E) regardless of the
// exact slot permutation. Wave-butterfly reduce D and divide by 16.
// ---------------------------------------------------------------------------
__global__ void count_kernel(const int* __restrict__ nb,
                             int* __restrict__ counts, int N) {
    const int lane = threadIdx.x & 31;
    const int node = blockIdx.x * (blockDim.x >> 5) + (threadIdx.x >> 5);
    if (node >= N) return;  // wave-uniform: active waves keep EXEC all-ones

    const int* nbi = nb + node * 16;
    int av[NEI];
#pragma unroll
    for (int k = 0; k < NEI; ++k) av[k] = nbi[k];

    // lanes 0..14 each own neighbor row j = lane; everyone else contributes 0
    const bool havelist = (lane < NEI);
    int am = 0;
    int rl[NEI];
    unsigned memb = 0;  // bit k: av[k] is adjacent to av[lane]
    if (havelist) {
        am = av[lane];
        const int* p = nb + am * 16;
#pragma unroll
        for (int l = 0; l < NEI; ++l) rl[l] = p[l];
#pragma unroll
        for (int k = 0; k < NEI; ++k) {
            int b = av[k];
            bool adj = false;
#pragma unroll
            for (int l = 0; l < NEI; ++l) adj |= (rl[l] == b);
            if (adj) memb |= (1u << k);
        }
    }

    v16h A;
    v16h B;
#pragma unroll
    for (int k = 0; k < 16; ++k) {
        bool e = havelist && (k < NEI) && (k > lane) &&
                 (am > node) && (av[k] > node) && ((memb >> k) & 1u);
        A[k] = (_Float16)(e ? 1.0f : 0.0f);
        B[k] = (_Float16)1.0f;
    }

    v8f c = {};
    // (neg_a, A, neg_b, B, c_mod, C, reuse_a, reuse_b)
    c = __builtin_amdgcn_wmma_f32_16x16x32_f16(false, A, false, B,
                                               (short)0, c, false, false);

    float s = c[0] + c[1] + c[2] + c[3] + c[4] + c[5] + c[6] + c[7];
#pragma unroll
    for (int m = 1; m < 32; m <<= 1) s += __shfl_xor(s, m, 32);
    if (lane == 0) counts[node] = (int)(s * (1.0f / 16.0f) + 0.5f);
}

// ---------------------------------------------------------------------------
// Kernel 3: single-block exclusive prefix sum over counts[0..n-1] -> offs.
// ---------------------------------------------------------------------------
__global__ void scan_kernel(const int* __restrict__ counts,
                            int* __restrict__ offs, int n) {
    __shared__ int sd[1024];
    __shared__ int carry;
    const int tid = threadIdx.x;
    if (tid == 0) carry = 0;
    __syncthreads();
    for (int base = 0; base < n; base += 1024) {
        int idx = base + tid;
        int v = (idx < n) ? counts[idx] : 0;
        sd[tid] = v;
        __syncthreads();
        for (int off = 1; off < 1024; off <<= 1) {
            int t = (tid >= off) ? sd[tid - off] : 0;
            __syncthreads();
            sd[tid] += t;
            __syncthreads();
        }
        if (idx < n) offs[idx] = carry + sd[tid] - v;  // exclusive
        int btot = sd[1023];
        __syncthreads();
        if (tid == 0) carry += btot;
        __syncthreads();
    }
    if (tid == 0) offs[n] = carry;
}

// ---------------------------------------------------------------------------
// Kernel 4: emit triangles. One thread per node; pairs are visited in (j,k)
// lex order == (a,b) lex order (neighbor lists ascending), and nodes ascend
// by offset, so the flat output is exactly the reference's sorted-unique list.
// ---------------------------------------------------------------------------
__global__ void emit_kernel(const int* __restrict__ nb,
                            const int* __restrict__ offs,
                            float* __restrict__ out, int N, int out_size) {
    const int i = blockIdx.x * blockDim.x + threadIdx.x;
    if (i >= N) return;
    const int* nbi = nb + i * 16;
    int pos = offs[i];
    for (int j = 0; j < NEI; ++j) {
        int a = nbi[j];
        if (a <= i) continue;
        const int* nba = nb + a * 16;
        int la[NEI];
#pragma unroll
        for (int l = 0; l < NEI; ++l) la[l] = nba[l];
        for (int k = j + 1; k < NEI; ++k) {
            int b = nbi[k];
            if (b <= i) continue;
            bool adj = false;
#pragma unroll
            for (int l = 0; l < NEI; ++l) adj |= (la[l] == b);
            if (adj) {
                size_t w = (size_t)pos * 3;
                if (w + 2 < (size_t)out_size) {
                    out[w + 0] = (float)i;
                    out[w + 1] = (float)a;
                    out[w + 2] = (float)b;
                }
                ++pos;
            }
        }
    }
}

extern "C" void kernel_launch(void* const* d_in, const int* in_sizes, int n_in,
                              void* d_out, int out_size, void* d_ws, size_t ws_size,
                              hipStream_t stream) {
    (void)n_in; (void)ws_size;
    const float* adj = (const float*)d_in[0];
    const int N = (int)(sqrt((double)in_sizes[0]) + 0.5);

    int* nb     = (int*)d_ws;          // N*16 ints
    int* counts = nb + (size_t)N * 16; // N ints
    int* offs   = counts + N;          // N+1 ints

    const int wavesPerBlock = 8;                     // 256 threads, wave32
    const int blocksWave = (N + wavesPerBlock - 1) / wavesPerBlock;

    extract_nb_kernel<<<blocksWave, 256, 0, stream>>>(adj, nb, N);
    count_kernel<<<blocksWave, 256, 0, stream>>>(nb, counts, N);
    scan_kernel<<<1, 1024, 0, stream>>>(counts, offs, N);
    emit_kernel<<<(N + 255) / 256, 256, 0, stream>>>(nb, offs, (float*)d_out,
                                                     N, out_size);
}
